// SoftHd_30399778521506
// MI455X (gfx1250) — matrix operand order repaired and checked by reading, not compiled, not run on hardware
//
#include <hip/hip_runtime.h>
#include <hip/hip_bf16.h>

// ---------------------------------------------------------------------------
// Soft Hausdorff, B=32, N=M=1024, D=256, fp32 in / fp32 out.
// Cross term via v_wmma_f32_16x16x32_bf16 (fp32 accum), norms in exact fp32.
// Pipelined: double-buffered LDS, global loads for step k+1 issued before the
// barrier, fragments hoisted. Waves tiled 4x2 over the 128x128 block so each
// wave computes 32 rows x 64 cols from 2 A-frags x 4 B-frags (8 WMMAs),
// cutting redundant LDS fragment traffic by a third vs a 1-D wave split.
// ---------------------------------------------------------------------------

typedef __attribute__((ext_vector_type(16))) __bf16 v16bf;
typedef __attribute__((ext_vector_type(4)))  __bf16 v4bf;
typedef __attribute__((ext_vector_type(8)))  float  v8f;
typedef __attribute__((ext_vector_type(4)))  float  v4fe;

#define B_  32
#define N_  1024
#define M_  1024
#define D_  256

#define LDSW 40          // ushorts per LDS row (80 bytes: 64B data + 16B pad)
#define INF_BITS 0x7F800000u

union FragU { uint4 q[2]; v16bf v; };

static __device__ __forceinline__ uint2 packbf4(float4 v) {
    // native RNE fp32 -> bf16 (v_cvt_pk_bf16_f32 class), 4 values -> 8 bytes
    v4fe vf = { v.x, v.y, v.z, v.w };
    union { v4bf b; uint2 u; } c;
    c.b = __builtin_convertvector(vf, v4bf);
    return c.u;
}

// ---------------------------------------------------------------------------
// Kernel 1: squared norms per point + seed min arrays with +inf.
// One wave (32 lanes) per 256-float row.
// ---------------------------------------------------------------------------
__global__ __launch_bounds__(256) void softhd_prep(
    const float* __restrict__ h1, const float* __restrict__ h2,
    float* __restrict__ sqn, float* __restrict__ sqm,
    unsigned* __restrict__ rowMin, unsigned* __restrict__ colMin)
{
    const int gwave = (blockIdx.x * 256 + threadIdx.x) >> 5;  // 0 .. B*(N+M)-1
    const int lane  = threadIdx.x & 31;
    const int NB = B_ * N_;

    const float* src;
    float* dst;
    unsigned* mins;
    int idx;
    if (gwave < NB) { idx = gwave;      src = h1 + (size_t)idx * D_; dst = sqn; mins = rowMin; }
    else            { idx = gwave - NB; src = h2 + (size_t)idx * D_; dst = sqm; mins = colMin; }

    float s = 0.0f;
#pragma unroll
    for (int i = 0; i < D_ / 32; ++i) {
        float v = src[lane + 32 * i];
        s += v * v;
    }
#pragma unroll
    for (int m = 16; m >= 1; m >>= 1) s += __shfl_xor(s, m, 32);

    if (lane == 0) { dst[idx] = s; mins[idx] = INF_BITS; }
}

// ---------------------------------------------------------------------------
// Kernel 2: 128x128 distance tile per block via bf16 WMMA, fused min-reduce.
// grid = (M/128, N/128, B), block = 256 (8 waves as 4 row-groups x 2
// col-groups). Wave (wr,wc) owns rows [wr*32, +32) x cols [wc*64, +64):
// 2 A-fragments x 4 B-fragments -> 8 accumulators. Double-buffered LDS.
// ---------------------------------------------------------------------------
__global__ __launch_bounds__(256) void softhd_main(
    const float* __restrict__ h1, const float* __restrict__ h2,
    const float* __restrict__ sqn, const float* __restrict__ sqm,
    unsigned* __restrict__ rowMin, unsigned* __restrict__ colMin)
{
    __shared__ __align__(16) unsigned short ldsA[2][128 * LDSW];
    __shared__ __align__(16) unsigned short ldsB[2][128 * LDSW];
    __shared__ unsigned ldsCol[128];

    const int tid  = threadIdx.x;
    const int b    = blockIdx.z;
    const int n0   = blockIdx.y * 128;
    const int m0   = blockIdx.x * 128;
    const int wv   = tid >> 5;
    const int wr   = wv & 3;      // row group: rows [wr*32, +32)
    const int wc   = wv >> 2;     // col group: cols [wc*64, +64)
    const int lane = tid & 31;
    const int lr   = lane & 15;   // col index within 16 / row-in-16
    const int kh   = lane >> 4;   // K-half per the ISA lane layout

    if (tid < 128) ldsCol[tid] = INF_BITS;

    const size_t baseA = ((size_t)b * N_ + n0) * D_;
    const size_t baseB = ((size_t)b * M_ + m0) * D_;

    // Per-thread staging coordinates: 128 rows x 8 float4 per operand slab.
    int srow[4], soff[4];
#pragma unroll
    for (int i = 0; i < 4; ++i) {
        int f   = tid + i * 256;
        int row = f >> 3;
        int kc  = f & 7;
        soff[i] = row * LDSW + kc * 4;   // ushort index in LDS
        srow[i] = row * D_ + kc * 4;     // float offset within slab
    }

    v8f acc[2][4] = {};
    float4 ra[4], rb[4];

    // ---- prologue: load + stage K-step 0 into buffer 0 ----
#pragma unroll
    for (int i = 0; i < 4; ++i) {
        ra[i] = *(const float4*)(h1 + baseA + srow[i]);
        rb[i] = *(const float4*)(h2 + baseB + srow[i]);
    }
#pragma unroll
    for (int i = 0; i < 4; ++i) {
        *(uint2*)&ldsA[0][soff[i]] = packbf4(ra[i]);
        *(uint2*)&ldsB[0][soff[i]] = packbf4(rb[i]);
    }

    for (int s = 0; s < 8; ++s) {
        const int cur = s & 1;

        // issue next slab's global loads before the barrier (latency hiding)
        if (s < 7) {
            const int k0n = (s + 1) * 32;
#pragma unroll
            for (int i = 0; i < 4; ++i) {
                ra[i] = *(const float4*)(h1 + baseA + srow[i] + k0n);
                rb[i] = *(const float4*)(h2 + baseB + srow[i] + k0n);
            }
        }

        __syncthreads();   // buffer `cur` fully staged

        // ---- fragments: 2x A (16x32) + 4x B (32x16) ----
        // A: elems 0..7  <- K = kh*8..+7      (bytes [kh*16, +16))
        //    elems 8..15 <- K = 16+kh*8..+7   (bytes [32+kh*16, +16))
        v16bf av[2];
#pragma unroll
        for (int ar = 0; ar < 2; ++ar) {
            const uint4* pa4 =
                (const uint4*)&ldsA[cur][(wr * 32 + ar * 16 + lr) * LDSW];
            FragU fa;
            fa.q[0] = pa4[kh];
            fa.q[1] = pa4[2 + kh];
            av[ar] = fa.v;
        }

        v16bf bv[4];
#pragma unroll
        for (int t = 0; t < 4; ++t) {
            // B: lane lr = column, elems 0..7 <- K kh*8..+7, 8..15 <- K 16+kh*8..+7
            const uint4* pb4 =
                (const uint4*)&ldsB[cur][(wc * 64 + t * 16 + lr) * LDSW];
            FragU fb;
            fb.q[0] = pb4[kh];
            fb.q[1] = pb4[2 + kh];
            bv[t] = fb.v;
        }

#pragma unroll
        for (int ar = 0; ar < 2; ++ar)
#pragma unroll
            for (int t = 0; t < 4; ++t)
                acc[ar][t] = __builtin_amdgcn_wmma_f32_16x16x32_bf16(
                    false, av[ar], false, bv[t], (short)0, acc[ar][t],
                    false, false);

        // convert + store next slab into the other buffer (protected by the
        // __syncthreads at the top of iteration s+1)
        if (s < 7) {
            const int nxt = cur ^ 1;
#pragma unroll
            for (int i = 0; i < 4; ++i) {
                *(uint2*)&ldsA[nxt][soff[i]] = packbf4(ra[i]);
                *(uint2*)&ldsB[nxt][soff[i]] = packbf4(rb[i]);
            }
        }
    }

    // ---- epilogue: dist = sq1 + sq2 - 2*cross, clamp, fused min-reduce ----
    // C layout: acc[ar][t] elem r -> row (wr*32 + ar*16 + kh*8 + r),
    //                                col (wc*64 + t*16 + lr).
    float sqn_r[2][8], rmin[2][8];
#pragma unroll
    for (int ar = 0; ar < 2; ++ar)
#pragma unroll
        for (int r = 0; r < 8; ++r) {
            sqn_r[ar][r] =
                sqn[(size_t)b * N_ + n0 + wr * 32 + ar * 16 + kh * 8 + r];
            rmin[ar][r] = 3.4e38f;
        }

#pragma unroll
    for (int t = 0; t < 4; ++t) {
        float sc   = sqm[(size_t)b * M_ + m0 + wc * 64 + t * 16 + lr];
        float cmin = 3.4e38f;
#pragma unroll
        for (int ar = 0; ar < 2; ++ar)
#pragma unroll
            for (int r = 0; r < 8; ++r) {
                float d = fmaxf(sqn_r[ar][r] + sc - 2.0f * acc[ar][t][r], 0.0f);
                rmin[ar][r] = fminf(rmin[ar][r], d);
                cmin        = fminf(cmin, d);
            }
        // column min across the two row-halves (lane L <-> L^16)
        cmin = fminf(cmin, __shfl_xor(cmin, 16, 32));
        if (kh == 0)
            atomicMin(&ldsCol[wc * 64 + t * 16 + lr], __float_as_uint(cmin));
    }

    // row min across the 16 columns held by lanes of the same half; the other
    // 64 columns of this row belong to the wc-partner wave (merged by the
    // global atomic below).
#pragma unroll
    for (int ar = 0; ar < 2; ++ar)
#pragma unroll
        for (int r = 0; r < 8; ++r) {
            float v = rmin[ar][r];
            v = fminf(v, __shfl_xor(v, 1, 32));
            v = fminf(v, __shfl_xor(v, 2, 32));
            v = fminf(v, __shfl_xor(v, 4, 32));
            v = fminf(v, __shfl_xor(v, 8, 32));
            if (lr == 0)
                atomicMin(&rowMin[(size_t)b * N_ + n0 + wr * 32 + ar * 16 +
                                  kh * 8 + r],
                          __float_as_uint(v));
        }

    __syncthreads();
    if (tid < 128)
        atomicMin(&colMin[(size_t)b * M_ + m0 + tid], ldsCol[tid]);
}

// ---------------------------------------------------------------------------
// Kernel 3: per-batch sum of row/col mins, divide by min(N,M).
// ---------------------------------------------------------------------------
__global__ __launch_bounds__(256) void softhd_finalize(
    const float* __restrict__ rowMin, const float* __restrict__ colMin,
    float* __restrict__ out)
{
    __shared__ float red[256];
    const int b   = blockIdx.x;
    const int tid = threadIdx.x;
    float s = 0.0f;
    for (int i = tid; i < N_; i += 256)
        s += rowMin[(size_t)b * N_ + i] + colMin[(size_t)b * M_ + i];
    red[tid] = s;
    __syncthreads();
    for (int off = 128; off > 0; off >>= 1) {
        if (tid < off) red[tid] += red[tid + off];
        __syncthreads();
    }
    if (tid == 0) out[b] = red[0] * (1.0f / (float)N_);
}

// ---------------------------------------------------------------------------
extern "C" void kernel_launch(void* const* d_in, const int* in_sizes, int n_in,
                              void* d_out, int out_size, void* d_ws, size_t ws_size,
                              hipStream_t stream)
{
    const float* h1 = (const float*)d_in[0];   // [B,N,D] fp32
    const float* h2 = (const float*)d_in[1];   // [B,M,D] fp32

    // workspace layout (floats): sqn[B*N] | sqm[B*M] | rowMin[B*N] | colMin[B*M]
    float*    ws     = (float*)d_ws;
    float*    sqn    = ws;
    float*    sqm    = ws + (size_t)B_ * N_;
    unsigned* rowMin = (unsigned*)(ws + 2u * B_ * N_);
    unsigned* colMin = (unsigned*)(ws + 3u * B_ * N_);

    // 1) norms + inf-seed: B*(N+M) waves, 8 waves/block
    const int totalWaves = B_ * (N_ + M_);
    softhd_prep<<<totalWaves / 8, 256, 0, stream>>>(h1, h2, sqn, sqm, rowMin, colMin);

    // 2) WMMA distance tiles + fused min reductions
    dim3 grid(M_ / 128, N_ / 128, B_);
    softhd_main<<<grid, 256, 0, stream>>>(h1, h2, sqn, sqm, rowMin, colMin);

    // 3) per-batch sums
    softhd_finalize<<<B_, 256, 0, stream>>>((const float*)rowMin, (const float*)colMin,
                                            (float*)d_out);
}
